// GraphSAGEModel_70806830841997
// MI455X (gfx1250) — matrix operand order repaired and checked by reading, not compiled, not run on hardware
//
#include <hip/hip_runtime.h>
#include <hip/hip_bf16.h>

typedef __attribute__((ext_vector_type(16))) _Float16 v16h;
typedef __attribute__((ext_vector_type(8)))  _Float16 v8h;
typedef __attribute__((ext_vector_type(8)))  float    v8f;

#define ATOMIC_ADD_F32(p, v) \
    __hip_atomic_fetch_add((p), (v), __ATOMIC_RELAXED, __HIP_MEMORY_SCOPE_AGENT)

// Fast reciprocal: single v_rcp_f32 instead of IEEE divide sequence.
__device__ __forceinline__ float fast_rcp(float x) {
    return __builtin_amdgcn_rcpf(x);
}

// -------------------------------------------------------------------------
// Packed f16 atomic add (GLOBAL_ATOMIC_PK_ADD_F16, no return), saddr form:
// uniform SGPR64 base + 32-bit VGPR byte offset + immediate offset.
// -------------------------------------------------------------------------
__device__ __forceinline__ void atomic_pk_add_f16(const _Float16* base,
                                                  unsigned voff_bytes,
                                                  unsigned int data, int ioff) {
    asm volatile("global_atomic_pk_add_f16 %0, %1, %2 offset:%c3"
                 :: "v"(voff_bytes), "v"(data), "s"(base), "i"(ioff)
                 : "memory");
}

// -------------------------------------------------------------------------
// Kernel 1: per-edge degree + layer-1 (2-dim) message accumulation
// -------------------------------------------------------------------------
__global__ void k_edge1(const int* __restrict__ src, const int* __restrict__ dst,
                        const int* __restrict__ um,  const int* __restrict__ vm,
                        float* __restrict__ deg, float* __restrict__ msg1, int E) {
    int e = blockIdx.x * blockDim.x + threadIdx.x;
    if (e >= E) return;
    unsigned s = (unsigned)src[e], d = (unsigned)dst[e];
    ATOMIC_ADD_F32(&deg[d], 1.0f);
    ATOMIC_ADD_F32(&msg1[2u * d + 0u], (float)um[s]);
    ATOMIC_ADD_F32(&msg1[2u * d + 1u], (float)vm[s]);
}

// -------------------------------------------------------------------------
// Kernel 2: swizzle w2_self / w2_neigh into WMMA B-fragment order, f16.
// Fragment layout (16-bit B, 32x16): lane n<16 holds column n for K=0..7,16..23,
// lane n+16 holds K=8..15,24..31.  Linear index:
//   ((colTile*2 + kChunk)*32 + lane)*16 + h ,  k = kChunk*32 + (lane>>4)*8 + (h&7) + (h>>3)*16
// -------------------------------------------------------------------------
__global__ void k_prep_b(const float* __restrict__ w2s, const float* __restrict__ w2n,
                         _Float16* __restrict__ bS, _Float16* __restrict__ bN) {
    int t = blockIdx.x * blockDim.x + threadIdx.x;   // 0..8191
    if (t >= 8192) return;
    int mat  = t >> 12;
    int idx  = t & 4095;
    int h    = idx & 15;
    int lane = (idx >> 4) & 31;
    int kc   = (idx >> 9) & 1;
    int ct   = idx >> 10;
    int k    = kc * 32 + (lane >> 4) * 8 + (h & 7) + (h >> 3) * 16;
    int col  = ct * 16 + (lane & 15);
    const float* w = mat ? w2n : w2s;
    _Float16*    o = mat ? bN  : bS;
    o[idx] = (_Float16)w[k * 64 + col];
}

// -------------------------------------------------------------------------
// Kernel 3: layer-1 node update (in=2 -> 64), writes h1 as f16; also graph
// node counts.
// -------------------------------------------------------------------------
__global__ void k_node1(const int* __restrict__ um, const int* __restrict__ vm,
                        const int* __restrict__ gid,
                        const float* __restrict__ deg, const float* __restrict__ msg1,
                        const float* __restrict__ w1s, const float* __restrict__ w1n,
                        const float* __restrict__ b1,
                        _Float16* __restrict__ h1, float* __restrict__ counts, int N) {
    int t = blockIdx.x * blockDim.x + threadIdx.x;
    if (t >= N * 64) return;
    unsigned n = (unsigned)t >> 6, j = (unsigned)t & 63u;
    float u = (float)um[n], v = (float)vm[n];
    float inv = fast_rcp(fmaxf(deg[n], 1.0f));
    float nb0 = msg1[2u * n + 0u] * inv;
    float nb1 = msg1[2u * n + 1u] * inv;
    float acc = u * w1s[j] + v * w1s[64u + j] + nb0 * w1n[j] + nb1 * w1n[64u + j] + b1[j];
    h1[(size_t)n * 64 + j] = (_Float16)fmaxf(acc, 0.0f);
    if (j == 0) ATOMIC_ADD_F32(&counts[(unsigned)gid[n]], 1.0f);
}

// -------------------------------------------------------------------------
// Kernel 4: layer-2 edge phase. 8 threads per edge, each moves 8 channels:
// one 16B vector gather from h1[src], four packed-f16 atomics into msg2[dst].
// Working set is L2-resident (h1 + msg2 = 25.6 MB << 192 MB).
// -------------------------------------------------------------------------
__global__ void k_edge2(const int* __restrict__ src, const int* __restrict__ dst,
                        const _Float16* __restrict__ h1,
                        _Float16* __restrict__ msg2, int E) {
    int t = blockIdx.x * blockDim.x + threadIdx.x;
    if (t >= E * 8) return;
    unsigned e = (unsigned)t >> 3;
    unsigned c = ((unsigned)t & 7u) * 8u;
    unsigned s = (unsigned)src[e], d = (unsigned)dst[e];
    union { v8h v; unsigned int u[4]; } x;
    x.v = *(const v8h*)(h1 + (size_t)s * 64 + c);
    unsigned voff = (d * 64u + c) * 2u;           // byte offset, < 2^31
    atomic_pk_add_f16(msg2, voff, x.u[0], 0);
    atomic_pk_add_f16(msg2, voff, x.u[1], 4);
    atomic_pk_add_f16(msg2, voff, x.u[2], 8);
    atomic_pk_add_f16(msg2, voff, x.u[3], 12);
}

// -------------------------------------------------------------------------
// Kernel 5: layer-2 node update via WMMA + fused bias/ReLU/graph-mean-pool.
// One wave per 16-node tile. A fragments follow the documented 16-bit A
// (16x32) layout: lane m(<16)/m+16 holds row m; halves 0..7 -> K = kb..kb+7,
// halves 8..15 -> K = kb+16..kb+23, kb = kChunk*32 + (lane>>4)*8.
// -------------------------------------------------------------------------
__global__ void k_sage2_wmma(const _Float16* __restrict__ h1,
                             const _Float16* __restrict__ msg2,
                             const float* __restrict__ deg,
                             const _Float16* __restrict__ bS,
                             const _Float16* __restrict__ bN,
                             const float* __restrict__ b2,
                             const int* __restrict__ gid,
                             float* __restrict__ pooled, int nTiles) {
    int wave = blockIdx.x * (blockDim.x >> 5) + (threadIdx.x >> 5);
    if (wave >= nTiles) return;                 // wave-uniform: EXEC stays full
    unsigned lane = threadIdx.x & 31u;
    unsigned m    = lane & 15u;
    unsigned hi   = lane >> 4;

    unsigned nodeA = (unsigned)wave * 16u + m;
    _Float16 inv = (_Float16)fast_rcp(fmaxf(deg[nodeA], 1.0f));
    v8h invv;
#pragma unroll
    for (int i = 0; i < 8; ++i) invv[i] = inv;

    v16h aS[2], aN[2];
#pragma unroll
    for (int kc = 0; kc < 2; ++kc) {
        unsigned kb = (unsigned)kc * 32u + hi * 8u;
        v8h slo = *(const v8h*)(h1   + (size_t)nodeA * 64 + kb);
        v8h shi = *(const v8h*)(h1   + (size_t)nodeA * 64 + kb + 16);
        v8h nlo = *(const v8h*)(msg2 + (size_t)nodeA * 64 + kb);
        v8h nhi = *(const v8h*)(msg2 + (size_t)nodeA * 64 + kb + 16);
        nlo *= invv;                             // v_pk_mul_f16
        nhi *= invv;
#pragma unroll
        for (int i = 0; i < 8; ++i) {
            aS[kc][i]     = slo[i];
            aS[kc][i + 8] = shi[i];
            aN[kc][i]     = nlo[i];
            aN[kc][i + 8] = nhi[i];
        }
    }

    unsigned g8[8];
#pragma unroll
    for (int r = 0; r < 8; ++r)
        g8[r] = (unsigned)gid[(unsigned)wave * 16u + (unsigned)r + hi * 8u];

#pragma unroll
    for (int ct = 0; ct < 4; ++ct) {
        v8f acc = {};
#pragma unroll
        for (int kc = 0; kc < 2; ++kc) {
            v16h fbS = *(const v16h*)(bS + ((size_t)(ct * 2 + kc) * 32 + lane) * 16);
            v16h fbN = *(const v16h*)(bN + ((size_t)(ct * 2 + kc) * 32 + lane) * 16);
            acc = __builtin_amdgcn_wmma_f32_16x16x32_f16(
                      false, aS[kc], false, fbS, (short)0, acc, false, false);
            acc = __builtin_amdgcn_wmma_f32_16x16x32_f16(
                      false, aN[kc], false, fbN, (short)0, acc, false, false);
        }
        unsigned j   = (unsigned)ct * 16u + m;   // output column
        float   bias = b2[j];
#pragma unroll
        for (int r = 0; r < 8; ++r) {
            float v = fmaxf(acc[r] + bias, 0.0f);
            ATOMIC_ADD_F32(&pooled[g8[r] * 64u + j], v);   // 32-bit offset, saddr form
        }
    }
}

// -------------------------------------------------------------------------
// Kernel 6: scorer MLP per graph: relu(mean @ ws1 + bs1) @ ws2 + bs2
// -------------------------------------------------------------------------
__global__ void k_scorer(const float* __restrict__ pooled, const float* __restrict__ counts,
                         const float* __restrict__ ws1, const float* __restrict__ bs1,
                         const float* __restrict__ ws2, const float* __restrict__ bs2,
                         float* __restrict__ out) {
    __shared__ float hg[64];
    __shared__ float red[64];
    unsigned g = blockIdx.x, j = threadIdx.x;
    float inv = fast_rcp(fmaxf(counts[g], 1.0f));
    hg[j] = pooled[(size_t)g * 64 + j] * inv;
    __syncthreads();
    float acc = bs1[j];
#pragma unroll 8
    for (int k = 0; k < 64; ++k) acc += hg[k] * ws1[k * 64 + j];
    red[j] = fmaxf(acc, 0.0f) * ws2[j];
    __syncthreads();
#pragma unroll
    for (int s = 32; s > 0; s >>= 1) {
        if (j < s) red[j] += red[j + s];
        __syncthreads();
    }
    if (j == 0) out[g] = red[0] + bs2[0];
}

// -------------------------------------------------------------------------
extern "C" void kernel_launch(void* const* d_in, const int* in_sizes, int n_in,
                              void* d_out, int out_size, void* d_ws, size_t ws_size,
                              hipStream_t stream) {
    const int*   src  = (const int*)d_in[0];
    const int*   dst  = (const int*)d_in[1];
    const int*   gid  = (const int*)d_in[2];
    const int*   um   = (const int*)d_in[3];
    const int*   vm   = (const int*)d_in[4];
    const float* w1s  = (const float*)d_in[5];
    const float* w1n  = (const float*)d_in[6];
    const float* b1   = (const float*)d_in[7];
    const float* w2s  = (const float*)d_in[8];
    const float* w2n  = (const float*)d_in[9];
    const float* b2   = (const float*)d_in[10];
    const float* ws1  = (const float*)d_in[11];
    const float* bs1  = (const float*)d_in[12];
    const float* ws2  = (const float*)d_in[13];
    const float* bs2  = (const float*)d_in[14];
    float*       out  = (float*)d_out;

    const int E = in_sizes[0];
    const int N = in_sizes[2];
    const int G = out_size;

    // Workspace layout (all 32B aligned for N=100000, G=1024)
    char* ws = (char*)d_ws;
    float*    deg    = (float*)ws;                    // N
    float*    msg1   = deg + N;                       // 2N
    float*    pooled = msg1 + 2 * (size_t)N;          // 64G
    float*    counts = pooled + 64 * (size_t)G;       // G
    _Float16* msg2   = (_Float16*)(counts + G);       // 64N halves
    _Float16* h1     = msg2 + 64 * (size_t)N;         // 64N halves
    _Float16* bS     = h1 + 64 * (size_t)N;           // 4096 halves
    _Float16* bN     = bS + 4096;                     // 4096 halves

    // Zero all accumulator regions in one memset (deg..counts f32 + msg2 f16)
    size_t zeroBytes = ((size_t)3 * N + 65 * (size_t)G) * 4 + (size_t)64 * N * 2;
    hipMemsetAsync(ws, 0, zeroBytes, stream);

    k_edge1<<<(E + 255) / 256, 256, 0, stream>>>(src, dst, um, vm, deg, msg1, E);
    k_prep_b<<<32, 256, 0, stream>>>(w2s, w2n, bS, bN);
    k_node1<<<((size_t)N * 64 + 255) / 256, 256, 0, stream>>>(
        um, vm, gid, deg, msg1, w1s, w1n, b1, h1, counts, N);
    k_edge2<<<((size_t)E * 8 + 255) / 256, 256, 0, stream>>>(src, dst, h1, msg2, E);

    int nTiles = (N + 15) / 16;                       // 6250 for N=100000
    k_sage2_wmma<<<(nTiles + 7) / 8, 256, 0, stream>>>(
        h1, msg2, deg, bS, bN, b2, gid, pooled, nTiles);

    k_scorer<<<G, 64, 0, stream>>>(pooled, counts, ws1, bs1, ws2, bs2, out);
}
